// ProcessHandBoardConv_30786325577966
// MI455X (gfx1250) — compile-verified, matrix operand order checked by reading, not run on hardware
//
#include <hip/hip_runtime.h>
#include <hip/hip_bf16.h>
#include <cstdint>
#include <cstddef>

// ---------------------------------------------------------------------------
// Types for CDNA5 WMMA (gfx1250, wave32)
// ---------------------------------------------------------------------------
typedef __attribute__((ext_vector_type(16))) _Float16 v16h;
typedef __attribute__((ext_vector_type(8)))  _Float16 v8h;
typedef __attribute__((ext_vector_type(8)))  float    v8f;

// ---------------------------------------------------------------------------
// Problem constants
// ---------------------------------------------------------------------------
constexpr int kB = 16, kM = 24, kCombos = 60;
constexpr int kBM    = kB * kM;          // 384
constexpr int kHands = kBM * kCombos;    // 23040 (divisible by 16 -> 1440 tiles)
constexpr int kCls   = 7463;
constexpr int kClsP  = 7472;             // padded to multiple of 16
constexpr int kNT    = kClsP / 16;       // 467 class tiles
constexpr int kRawLen = 256;             // 16x16 raw features per hand
constexpr int kHLen   = 512;             // flattened h per hand
constexpr int kOFeat  = 15360;           // 60*256 per (b,m)

// Workspace layout (bytes), 256B aligned blocks. Total ~60.3 MB.
constexpr size_t alignUp256(size_t x) { return (x + 255) & ~size_t(255); }
constexpr size_t OFF_H    = 0;                                                    // f16 [23040][512]
constexpr size_t OFF_RAW  = alignUp256(OFF_H    + (size_t)kHands * kHLen  * 2);   // f16 [23040][256]
constexpr size_t OFF_WC   = alignUp256(OFF_RAW  + (size_t)kHands * kRawLen * 2);  // f16 [7472][512]
constexpr size_t OFF_BC   = alignUp256(OFF_WC   + (size_t)kClsP  * kHLen  * 2);   // f32 [7472]
constexpr size_t OFF_WO1  = alignUp256(OFF_BC   + (size_t)kClsP  * 4);            // f16 [512][15360]
constexpr size_t OFF_AMAX = alignUp256(OFF_WO1  + (size_t)kHLen  * kOFeat * 2);   // i32 [23040]
constexpr size_t OFF_O1   = alignUp256(OFF_AMAX + (size_t)kHands * 4);            // f32 [384][512]
constexpr size_t OFF_O2   = alignUp256(OFF_O1   + (size_t)kBM * 512 * 4);         // f32 [384][256]

// Omaha combo tables: choose 2 of hole cards {0..3} x choose 3 of board {4..8}
__device__ const int kPair[6][2]  = {{0,1},{0,2},{0,3},{1,2},{1,3},{2,3}};
__device__ const int kTrip[10][3] = {{4,5,6},{4,5,7},{4,5,8},{4,6,7},{4,6,8},
                                     {4,7,8},{5,6,7},{5,6,8},{5,7,8},{6,7,8}};

// ---------------------------------------------------------------------------
// Load one lane's 16-half WMMA fragment slice: two contiguous 8-half runs
// (k offsets [0..7] and [16..23] relative to the lane's k-base).
// Works for both global and LDS pointers (16-byte aligned).
// ---------------------------------------------------------------------------
__device__ __forceinline__ v16h load_frag16(const _Float16* p) {
    v8h lo = *(const v8h*)(p);
    v8h hi = *(const v8h*)(p + 16);
    v16h r;
#pragma unroll
    for (int i = 0; i < 8; ++i) { r[i] = lo[i]; r[i + 8] = hi[i]; }
    return r;
}

// Async global -> LDS 16-byte copy (CDNA5 GLOBAL_LOAD_ASYNC_TO_LDS_B128,
// GV mode: per-lane 64-bit global address, per-lane LDS byte offset).
// Tracked by ASYNCcnt; consumer must s_wait_asynccnt before reading LDS.
__device__ __forceinline__ void async_copy_b128(unsigned lds_off, const void* gptr) {
    unsigned long long ga = (unsigned long long)(uintptr_t)gptr;
    asm volatile("global_load_async_to_lds_b128 %0, %1, off"
                 :: "v"(lds_off), "v"(ga) : "memory");
}
__device__ __forceinline__ void wait_asynccnt0() {
    asm volatile("s_wait_asynccnt 0x0" ::: "memory");
}

// ===========================================================================
// Kernel 1: per-hand features + per-row MLP.
// thread = (hand, row o). raw row o only feeds h1 row o -> no cross-row deps.
// ===========================================================================
__global__ void __launch_bounds__(256)
feat_kernel(const int* __restrict__ x,
            const float* __restrict__ suit_w, const float* __restrict__ suit_b,
            const float* __restrict__ rank_w, const float* __restrict__ rank_b,
            const float* __restrict__ bn_s_g, const float* __restrict__ bn_s_b,
            const float* __restrict__ bn_s_m, const float* __restrict__ bn_s_v,
            const float* __restrict__ bn_r_g, const float* __restrict__ bn_r_b,
            const float* __restrict__ bn_r_m, const float* __restrict__ bn_r_v,
            const float* __restrict__ w1, const float* __restrict__ b1,
            const float* __restrict__ w2, const float* __restrict__ b2,
            _Float16* __restrict__ raw16, _Float16* __restrict__ h16) {
    __shared__ float s_w1[32 * 16];
    __shared__ float s_w2[32 * 32];
    __shared__ float s_b1[32], s_b2[32];
    __shared__ float s_rw[16 * 5 * 5];
    __shared__ float s_sw[16 * 5];
    __shared__ float s_rinv[16], s_radd[16], s_sinv[16], s_sadd[16];
    __shared__ float s_rbias[16], s_sbias[16];

    const int tid = threadIdx.x;
    for (int i = tid; i < 512;  i += 256) s_w1[i] = w1[i];
    for (int i = tid; i < 1024; i += 256) s_w2[i] = w2[i];
    for (int i = tid; i < 400;  i += 256) s_rw[i] = rank_w[i];
    if (tid < 80) s_sw[tid] = suit_w[tid];          // suit_w[16][5][1] flat
    if (tid < 32) { s_b1[tid] = b1[tid]; s_b2[tid] = b2[tid]; }
    if (tid < 16) {
        float ri = bn_r_g[tid] * rsqrtf(bn_r_v[tid] + 1e-5f);
        s_rinv[tid] = ri; s_radd[tid] = bn_r_b[tid] - bn_r_m[tid] * ri;
        float si = bn_s_g[tid] * rsqrtf(bn_s_v[tid] + 1e-5f);
        s_sinv[tid] = si; s_sadd[tid] = bn_s_b[tid] - bn_s_m[tid] * si;
        s_rbias[tid] = rank_b[tid]; s_sbias[tid] = suit_b[tid];
    }
    __syncthreads();

    const int idx  = blockIdx.x * 256 + tid;   // exactly kHands*16 threads
    const int hand = idx >> 4;
    const int o    = idx & 15;
    const int bm   = hand / 60;
    const int cmb  = hand % 60;
    const int* xb  = x + bm * 18;              // alternating rank/suit, 9 cards

    int cards[5];
    cards[0] = kPair[cmb / 10][0]; cards[1] = kPair[cmb / 10][1];
    cards[2] = kTrip[cmb % 10][0]; cards[3] = kTrip[cmb % 10][1]; cards[4] = kTrip[cmb % 10][2];
    int rk[5], st[5];
#pragma unroll
    for (int c = 0; c < 5; ++c) { rk[c] = xb[cards[c] * 2]; st[c] = xb[cards[c] * 2 + 1]; }

    float row[16];
    const float rinv = s_rinv[o], radd = s_radd[o];
    const float sinv = s_sinv[o], sadd = s_sadd[o];
    // rank conv (k=5, valid, length 15 -> 11): nonzero iff rank[c]-p in [0,5)
#pragma unroll
    for (int p = 0; p < 11; ++p) {
        float a = s_rbias[o];
#pragma unroll
        for (int c = 0; c < 5; ++c) {
            int d = rk[c] - p;
            if (d >= 0 && d < 5) a += s_rw[(o * 5 + c) * 5 + d];
        }
        a = a * rinv + radd;
        row[p] = a > 0.f ? a : 0.f;
    }
    // suit 1x1 conv over one-hot length 5 (suits are 1..4)
#pragma unroll
    for (int l = 0; l < 5; ++l) {
        float a = s_sbias[o];
#pragma unroll
        for (int c = 0; c < 5; ++c)
            if (st[c] == l) a += s_sw[o * 5 + c];
        a = a * sinv + sadd;
        row[11 + l] = a > 0.f ? a : 0.f;
    }

    _Float16* rp = raw16 + (size_t)hand * kRawLen + o * 16;
#pragma unroll
    for (int k = 0; k < 16; ++k) rp[k] = (_Float16)row[k];

    // h1 = relu(row @ w1.T + b1)  [32]
    float h1[32];
#pragma unroll
    for (int j = 0; j < 32; ++j) {
        float a = s_b1[j];
#pragma unroll
        for (int k = 0; k < 16; ++k) a += row[k] * s_w1[j * 16 + k];
        h1[j] = a > 0.f ? a : 0.f;
    }
    // h2 = relu(h1 @ w2.T + b2)  [32] -> h flat index o*32+j
    _Float16* hp = h16 + (size_t)hand * kHLen + o * 32;
#pragma unroll
    for (int j = 0; j < 32; ++j) {
        float a = s_b2[j];
#pragma unroll
        for (int k = 0; k < 32; ++k) a += h1[k] * s_w2[j * 32 + k];
        hp[j] = (_Float16)(a > 0.f ? a : 0.f);
    }
}

// ===========================================================================
// Kernel 2: weight prep. wc -> f16 padded [7472][512] (pad bias = -3e38 so
// padded classes never win argmax); wo1 -> f16.
// ===========================================================================
__global__ void prep_kernel(const float* __restrict__ wc, const float* __restrict__ bc,
                            const float* __restrict__ wo1,
                            _Float16* __restrict__ wc16, float* __restrict__ bcp,
                            _Float16* __restrict__ wo116) {
    const size_t tid0   = (size_t)blockIdx.x * blockDim.x + threadIdx.x;
    const size_t stride = (size_t)gridDim.x * blockDim.x;
    for (size_t t = tid0; t < (size_t)kClsP * kHLen; t += stride) {
        int cls = (int)(t / kHLen), k = (int)(t % kHLen);
        wc16[t] = (_Float16)(cls < kCls ? wc[(size_t)cls * kHLen + k] : 0.f);
    }
    for (size_t t = tid0; t < (size_t)kClsP; t += stride)
        bcp[t] = (t < (size_t)kCls) ? bc[t] : -3.0e38f;
    for (size_t t = tid0; t < (size_t)kHLen * kOFeat; t += stride)
        wo116[t] = (_Float16)wo1[t];
}

// ===========================================================================
// Kernel 3: logits GEMM [23040x512]@[512x7472] fused with per-hand argmax.
// One wave per 16-hand row tile, 8 waves/block. All 8 waves share each class
// tile's B (16KB) via double-buffered LDS. Staging uses CDNA5 async
// global->LDS copies (ASYNCcnt): issue for tile ct+1, run 16 WMMAs on tile
// ct (from LDS, DScnt only), s_wait_asynccnt 0, barrier. LDS rows padded to
// 520 halves so each lane's ds_load_b128 hits its own 4 banks.
// ===========================================================================
__global__ void __launch_bounds__(256)
cls_argmax_kernel(const _Float16* __restrict__ h16, const _Float16* __restrict__ wc16,
                  const float* __restrict__ bcp, int* __restrict__ amax) {
    __shared__ alignas(16) _Float16 sB[2][16][520];   // 2 x 16 classes x (512+8 pad)

    const int tid  = threadIdx.x;
    const int lane = tid & 31;
    const int wid  = tid >> 5;
    const int tile = blockIdx.x * 8 + wid;        // 0..1439, exact
    const int r    = lane & 15;                   // A row within tile
    const int kb   = (lane < 16) ? 0 : 8;         // f16 WMMA lane k-base

    // Staging geometry: 16KB tile = 1024 x 16B chunks; 4 chunks per thread.
    int srow[4], scol[4];
    unsigned ldsOff[2][4];
#pragma unroll
    for (int i = 0; i < 4; ++i) {
        int c = tid + i * 256;
        srow[i] = c >> 6;        // class row 0..15
        scol[i] = c & 63;        // 16B chunk within the 512-half row
        ldsOff[0][i] = (unsigned)(uintptr_t)(&sB[0][srow[i]][scol[i] * 8]);
        ldsOff[1][i] = (unsigned)(uintptr_t)(&sB[1][srow[i]][scol[i] * 8]);
    }

    // Preload A (16 hands x 512) into 128 VGPRs.
    const _Float16* ap = h16 + (size_t)(tile * 16 + r) * kHLen + kb;
    v16h A[16];
#pragma unroll
    for (int ks = 0; ks < 16; ++ks) A[ks] = load_frag16(ap + ks * 32);

    float mv[8]; int mi[8];
#pragma unroll
    for (int g = 0; g < 8; ++g) { mv[g] = -3.4e38f; mi[g] = 0; }

    const int n = lane & 15;                      // B column = class within tile

    // Async-stage class tile 0.
#pragma unroll
    for (int i = 0; i < 4; ++i)
        async_copy_b128(ldsOff[0][i],
                        wc16 + (size_t)srow[i] * kHLen + scol[i] * 8);
    wait_asynccnt0();
    __syncthreads();

    int buf = 0;
    for (int ct = 0; ct < kNT; ++ct, buf ^= 1) {
        // Fire async copies for tile ct+1 into the other buffer; they overlap
        // all 16 WMMAs below and never touch VGPRs or DScnt.
        const bool hasNext = (ct + 1 < kNT);
        if (hasNext) {
#pragma unroll
            for (int i = 0; i < 4; ++i)
                async_copy_b128(ldsOff[buf ^ 1][i],
                                wc16 + ((size_t)(ct + 1) * 16 + srow[i]) * kHLen + scol[i] * 8);
        }
        if (ct + 2 < kNT) {
#pragma unroll
            for (int i = 0; i < 4; ++i)
                __builtin_prefetch(wc16 + ((size_t)(ct + 2) * 16 + srow[i]) * kHLen + scol[i] * 8, 0, 2);
        }

        // 16 WMMAs on tile ct from LDS, B fragment pipelined one step ahead.
        const _Float16* lb = &sB[buf][n][0];
        v8f acc = {};
        v16h Bcur = load_frag16(lb + kb);
#pragma unroll
        for (int ks = 0; ks < 16; ++ks) {
            v16h Bnext = (ks < 15) ? load_frag16(lb + (ks + 1) * 32 + kb) : Bcur;
            acc = __builtin_amdgcn_wmma_f32_16x16x32_f16(
                false, A[ks], false, Bcur, (short)0, acc, false, false);
            Bcur = Bnext;
        }

        const float bias = bcp[ct * 16 + n];
        const int   cidx = ct * 16 + n;
#pragma unroll
        for (int g = 0; g < 8; ++g) {
            float v = acc[g] + bias;
            if (v > mv[g]) { mv[g] = v; mi[g] = cidx; }  // strict > keeps first max
        }

        // Our async copies for ct+1 must be LDS-resident before the barrier.
        wait_asynccnt0();
        __syncthreads();
    }

    // Reduce (max, smaller-idx-on-tie) across the 16 lanes sharing each row group.
#pragma unroll
    for (int m = 1; m <= 8; m <<= 1) {
#pragma unroll
        for (int g = 0; g < 8; ++g) {
            float ov = __shfl_xor(mv[g], m, 32);
            int   oi = __shfl_xor(mi[g], m, 32);
            if (ov > mv[g] || (ov == mv[g] && oi < mi[g])) { mv[g] = ov; mi[g] = oi; }
        }
    }
    if ((lane & 15) == 0) {
        const int mbase = (lane >> 4) * 8;        // lanes 0..15 -> rows 0..7, 16..31 -> 8..15
#pragma unroll
        for (int g = 0; g < 8; ++g) amax[tile * 16 + mbase + g] = mi[g];
    }
}

// ===========================================================================
// Kernel 4: best = min over 60 combos; write out[...,127] as float.
// ===========================================================================
__global__ void best_kernel(const int* __restrict__ amax, float* __restrict__ out) {
    int bm = blockIdx.x * blockDim.x + threadIdx.x;
    if (bm >= kBM) return;
    int mn = 0x7fffffff;
    for (int h = 0; h < kCombos; ++h) {
        int v = amax[bm * kCombos + h];
        mn = v < mn ? v : mn;
    }
    out[bm * 128 + 127] = (float)mn;
}

// ===========================================================================
// Kernel 5: o1 = relu(raw_flat[384,15360] @ wo1.T + bo1) via WMMA.
// One wave = one 16-col tile x FOUR 16-row tiles -> each loaded B fragment
// feeds 4 independent WMMAs (4x L2-traffic reuse, latency hidden by MACs).
// 32 col tiles x 6 row groups = 192 waves.
// ===========================================================================
__global__ void __launch_bounds__(256)
o1_kernel(const _Float16* __restrict__ raw16, const _Float16* __restrict__ wo116,
          const float* __restrict__ bo1, float* __restrict__ o1) {
    const int lane = threadIdx.x & 31;
    const int wid  = threadIdx.x >> 5;
    const int wg   = blockIdx.x * 8 + wid;        // 0..191, exact
    const int ct   = wg / 6;                      // col tile 0..31
    const int rtg  = wg % 6;                      // row-tile group 0..5 (4 tiles each)
    const int r    = lane & 15;
    const int kb   = (lane < 16) ? 0 : 8;

    const _Float16* bp  = wo116 + (size_t)(ct * 16 + r) * kOFeat + kb;  // col n == r
    const _Float16* ap0 = raw16 + (size_t)(rtg * 64 + r) * kOFeat + kb;

    v8f acc[4] = {{}, {}, {}, {}};
#pragma unroll 2
    for (int ks = 0; ks < kOFeat / 32; ++ks) {
        v16h Bv = load_frag16(bp + ks * 32);
#pragma unroll
        for (int t = 0; t < 4; ++t) {
            v16h Av = load_frag16(ap0 + (size_t)t * 16 * kOFeat + ks * 32);
            acc[t] = __builtin_amdgcn_wmma_f32_16x16x32_f16(
                false, Av, false, Bv, (short)0, acc[t], false, false);
        }
    }
    const float bias  = bo1[ct * 16 + (lane & 15)];
    const int   mbase = (lane >> 4) * 8;
#pragma unroll
    for (int t = 0; t < 4; ++t) {
        const int rt = rtg * 4 + t;
#pragma unroll
        for (int g = 0; g < 8; ++g) {
            float v = acc[t][g] + bias;
            v = v > 0.f ? v : 0.f;
            o1[(size_t)(rt * 16 + mbase + g) * 512 + ct * 16 + (lane & 15)] = v;
        }
    }
}

// ===========================================================================
// Kernels 6/7: tiny tail layers (scalar f32 is plenty: <0.2 GFLOP).
// ===========================================================================
__global__ void o2_kernel(const float* __restrict__ o1, const float* __restrict__ wo2,
                          const float* __restrict__ bo2, float* __restrict__ o2) {
    int idx = blockIdx.x * blockDim.x + threadIdx.x;   // 384*256 exact
    int rrow = idx / 256, j = idx % 256;
    float a = bo2[j];
    const float* op = o1 + (size_t)rrow * 512;
    const float* wp = wo2 + (size_t)j * 512;
    for (int k = 0; k < 512; ++k) a += op[k] * wp[k];
    o2[(size_t)rrow * 256 + j] = a > 0.f ? a : 0.f;
}

__global__ void o3_kernel(const float* __restrict__ o2, const float* __restrict__ wo3,
                          const float* __restrict__ bo3, float* __restrict__ out) {
    int idx = blockIdx.x * blockDim.x + threadIdx.x;
    if (idx >= kBM * 127) return;
    int rrow = idx / 127, j = idx % 127;
    float a = bo3[j];
    const float* op = o2 + (size_t)rrow * 256;
    const float* wp = wo3 + (size_t)j * 256;
    for (int k = 0; k < 256; ++k) a += op[k] * wp[k];
    out[rrow * 128 + j] = a > 0.f ? a : 0.f;
}

// ===========================================================================
// Host launcher
// ===========================================================================
extern "C" void kernel_launch(void* const* d_in, const int* in_sizes, int n_in,
                              void* d_out, int out_size, void* d_ws, size_t ws_size,
                              hipStream_t stream) {
    // setup_inputs() dict order:
    const int*   x       = (const int*)  d_in[0];
    const float* suit_w  = (const float*)d_in[1];
    const float* suit_b  = (const float*)d_in[2];
    const float* rank_w  = (const float*)d_in[3];
    const float* rank_b  = (const float*)d_in[4];
    const float* bn_s_g  = (const float*)d_in[5];
    const float* bn_s_b  = (const float*)d_in[6];
    const float* bn_s_m  = (const float*)d_in[7];
    const float* bn_s_v  = (const float*)d_in[8];
    const float* bn_r_g  = (const float*)d_in[9];
    const float* bn_r_b  = (const float*)d_in[10];
    const float* bn_r_m  = (const float*)d_in[11];
    const float* bn_r_v  = (const float*)d_in[12];
    const float* w1      = (const float*)d_in[13];
    const float* b1      = (const float*)d_in[14];
    const float* w2      = (const float*)d_in[15];
    const float* b2      = (const float*)d_in[16];
    const float* wc      = (const float*)d_in[17];
    const float* bc      = (const float*)d_in[18];
    const float* wo1     = (const float*)d_in[19];
    const float* bo1     = (const float*)d_in[20];
    const float* wo2     = (const float*)d_in[21];
    const float* bo2     = (const float*)d_in[22];
    const float* wo3     = (const float*)d_in[23];
    const float* bo3     = (const float*)d_in[24];
    float* out = (float*)d_out;

    char* ws = (char*)d_ws;
    _Float16* h16    = (_Float16*)(ws + OFF_H);
    _Float16* raw16  = (_Float16*)(ws + OFF_RAW);
    _Float16* wc16   = (_Float16*)(ws + OFF_WC);
    float*    bcp    = (float*)   (ws + OFF_BC);
    _Float16* wo116  = (_Float16*)(ws + OFF_WO1);
    int*      amax   = (int*)     (ws + OFF_AMAX);
    float*    o1     = (float*)   (ws + OFF_O1);
    float*    o2     = (float*)   (ws + OFF_O2);

    // 1) features + per-hand MLP -> raw_f16, h_f16   (23040*16 threads)
    feat_kernel<<<kHands * 16 / 256, 256, 0, stream>>>(
        x, suit_w, suit_b, rank_w, rank_b,
        bn_s_g, bn_s_b, bn_s_m, bn_s_v,
        bn_r_g, bn_r_b, bn_r_m, bn_r_v,
        w1, b1, w2, b2, raw16, h16);

    // 2) weight conversion / padding
    prep_kernel<<<2048, 256, 0, stream>>>(wc, bc, wo1, wc16, bcp, wo116);

    // 3) big GEMM + fused argmax (1440 waves, 8 waves/block, async LDS-staged B)
    cls_argmax_kernel<<<1440 / 8, 256, 0, stream>>>(h16, wc16, bcp, amax);

    // 4) min over combos -> out[...,127]
    best_kernel<<<2, 256, 0, stream>>>(amax, out);

    // 5) raw head: o1 via WMMA (192 waves, 4 row tiles each)
    o1_kernel<<<192 / 8, 256, 0, stream>>>(raw16, wo116, bo1, o1);

    // 6,7) small tail layers; o3 writes out[...,0:127]
    o2_kernel<<<kBM * 256 / 256, 256, 0, stream>>>(o1, wo2, bo2, o2);
    o3_kernel<<<(kBM * 127 + 255) / 256, 256, 0, stream>>>(o2, wo3, bo3, out);
}
